// HomoGNN_66374424592515
// MI455X (gfx1250) — matrix operand order repaired and checked by reading, not compiled, not run on hardware
//
#include <hip/hip_runtime.h>
#include <hip/hip_bf16.h>
#include <stdint.h>

// Problem constants (match reference)
#define HCC   292
#define HEADS 4
#define CDIM  73
#define K2PAD 320      // 292 padded to multiple of 32
#define NPADR 304      // 292 padded to multiple of 16 (output cols / weight rows)

typedef __attribute__((ext_vector_type(16))) _Float16 v16h;
typedef __attribute__((ext_vector_type(8)))  _Float16 v8h;
typedef __attribute__((ext_vector_type(8)))  float    v8f;

// ---- order-preserving float <-> uint key (for atomic u32 max as float max) ----
__device__ __forceinline__ unsigned fkey(float f) {
  unsigned u = __float_as_uint(f);
  return (u & 0x80000000u) ? ~u : (u | 0x80000000u);
}
__device__ __forceinline__ float funkey(unsigned k) {
  unsigned u = (k & 0x80000000u) ? (k & 0x7fffffffu) : ~k;
  return __uint_as_float(u);
}

// ---------------- utility kernels ----------------
__global__ void k_zero_u32(unsigned* __restrict__ p, long n) {
  long i = (long)blockIdx.x * blockDim.x + threadIdx.x;
  if (i < n) p[i] = 0u;
}

__global__ void k_cvt_f16(const float* __restrict__ x, _Float16* __restrict__ y, long n) {
  long i = (long)blockIdx.x * blockDim.x + threadIdx.x;
  if (i < n) y[i] = (_Float16)x[i];
}

// W [rows x k] f32 row-major  ->  Wh [NPADR x kpad] f16, zero padded
__global__ void k_cvt_w(const float* __restrict__ W, _Float16* __restrict__ Wh,
                        int rows, int k, int kpad) {
  long i = (long)blockIdx.x * blockDim.x + threadIdx.x;
  long tot = (long)NPADR * kpad;
  if (i >= tot) return;
  int r = (int)(i / kpad), c = (int)(i % kpad);
  Wh[i] = (r < rows && c < k) ? (_Float16)W[(long)r * k + c] : (_Float16)0.0f;
}

// zero padding columns [292, 320) of Ah[N][K2PAD]
__global__ void k_pad_ah(_Float16* __restrict__ Ah, long n) {
  long i = (long)blockIdx.x * blockDim.x + threadIdx.x;
  long tot = n * (K2PAD - HCC);
  if (i >= tot) return;
  long row = i / (K2PAD - HCC);
  int  j   = (int)(i % (K2PAD - HCC));
  Ah[row * K2PAD + HCC + j] = (_Float16)0.0f;
}

// ---------------- WMMA GEMM: Y[M x ncols] = A(f16,[M x K]) * Wh^T + bias ----------------
// Wh is f16 [NPADR x K] row-major (row n = output column n of Y).
// One wave computes one 16x16 tile; ISA-exact fragment layouts (wave32).
__global__ void k_gemm_wmma(const _Float16* __restrict__ A, const _Float16* __restrict__ Wh,
                            const float* __restrict__ bias, float* __restrict__ Y,
                            int M, int K, int ncols) {
  const int ntiles = NPADR / 16; // 19
  int wid  = ((int)blockIdx.x * (int)blockDim.x + (int)threadIdx.x) >> 5;
  int lane = threadIdx.x & 31;
  int mt = wid / ntiles, nt = wid % ntiles;
  if (mt * 16 >= M) return;
  int l15  = lane & 15;
  int mrow = mt * 16 + l15;          // A-frag: row per lane
  int ncol = nt * 16 + l15;          // B-frag: column per lane
  int hiA  = (lane & 16) ? 8  : 0;   // A-frag K sub-base for upper half-wave
  int hiB  = (lane & 16) ? 8  : 0;   // B-frag K sub-base for upper half-wave
  const _Float16* ar = A  + (size_t)mrow * K;
  const _Float16* wr = Wh + (size_t)ncol * K;
  v8f acc = {};
  for (int k0 = 0; k0 < K; k0 += 32) {
    union { v16h v; v8h h[2]; } a;
    a.h[0] = *(const v8h*)(ar + k0 + hiA);        // halves 0..7  -> K = base..base+7
    a.h[1] = *(const v8h*)(ar + k0 + hiA + 16);   // halves 8..15 -> K = base+16..base+23
    union { v16h v; v8h h[2]; } b;
    b.h[0] = *(const v8h*)(wr + k0 + hiB);        // halves 0..7  -> K = base..base+7
    b.h[1] = *(const v8h*)(wr + k0 + hiB + 16);   // halves 8..15 -> K = base+16..base+23
    acc = __builtin_amdgcn_wmma_f32_16x16x32_f16(false, a.v, false, b.v,
                                                 (short)0, acc, false, false);
  }
  if (ncol < ncols) {
    float bv = bias[ncol];
    int rbase = mt * 16 + ((lane & 16) ? 8 : 0);  // D: VGPR r -> row rbase+r
    #pragma unroll
    for (int r = 0; r < 8; ++r)
      Y[(size_t)(rbase + r) * ncols + ncol] = acc[r] + bv;
  }
}

// ---------------- GATv2 edge kernels ----------------
// s[e,h] = sum_c leakyrelu(xl[src,h,c]+xr[dst,h,c], 0.2)*att[h,c]; atomic max into mkey[dst,h]
__global__ void k_edge_score(const int* __restrict__ ei, const float* __restrict__ xl,
                             const float* __restrict__ xr, const float* __restrict__ att,
                             float* __restrict__ s, unsigned* __restrict__ mkey, long E) {
  long t = (long)blockIdx.x * blockDim.x + threadIdx.x;
  if (t >= E * HEADS) return;
  long e = t >> 2; int h = (int)(t & 3);
  int src = ei[e], dst = ei[E + e];
  const float* pl = xl + (size_t)src * HCC + h * CDIM;
  const float* pr = xr + (size_t)dst * HCC + h * CDIM;
  const float* pa = att + h * CDIM;
  float acc = 0.f;
  for (int c = 0; c < CDIM; ++c) {
    float f = pl[c] + pr[c];
    f = (f > 0.f) ? f : 0.2f * f;
    acc += f * pa[c];
  }
  s[t] = acc;
  atomicMax(mkey + (size_t)dst * HEADS + h, fkey(acc));
}

// ex = exp(s - m[dst]); s <- ex; den[dst,h] += ex
__global__ void k_edge_exp(const int* __restrict__ ei, float* __restrict__ s,
                           const unsigned* __restrict__ mkey, float* __restrict__ den, long E) {
  long t = (long)blockIdx.x * blockDim.x + threadIdx.x;
  if (t >= E * HEADS) return;
  long e = t >> 2; int h = (int)(t & 3);
  int dst = ei[E + e];
  float mm = funkey(mkey[(size_t)dst * HEADS + h]);
  float ex = __expf(s[t] - mm);
  s[t] = ex;
  atomicAdd(den + (size_t)dst * HEADS + h, ex);
}

// out[dst,:] += alpha[e,h(ch)] * xl[src,:]   (one wave per edge; lanes stride channels)
__global__ void k_edge_aggr(const int* __restrict__ ei, const float* __restrict__ xl,
                            const float* __restrict__ s, const float* __restrict__ den,
                            float* __restrict__ out, long E) {
  long w = ((long)blockIdx.x * blockDim.x + threadIdx.x) >> 5;
  int lane = threadIdx.x & 31;
  if (w >= E) return;
  int src = ei[w], dst = ei[E + w];
  const float* ps = s   + (size_t)w * HEADS;
  const float* pd = den + (size_t)dst * HEADS;
  float a0 = ps[0] / (pd[0] + 1e-16f);
  float a1 = ps[1] / (pd[1] + 1e-16f);
  float a2 = ps[2] / (pd[2] + 1e-16f);
  float a3 = ps[3] / (pd[3] + 1e-16f);
  const float* px = xl + (size_t)src * HCC;
  float* po = out + (size_t)dst * HCC;
  __builtin_prefetch(px, 0, 0);
  for (int ch = lane; ch < HCC; ch += 32) {
    float al = (ch < CDIM) ? a0 : (ch < 2 * CDIM) ? a1 : (ch < 3 * CDIM) ? a2 : a3;
    atomicAdd(po + ch, al * px[ch]);
  }
}

// ---------------- BatchNorm ----------------
#define BN_ROWS 64
__global__ void k_bn_stats(const float* __restrict__ h, float* __restrict__ gsum,
                           float* __restrict__ gsq, long n) {
  __shared__ float ls[HCC];
  __shared__ float lq[HCC];
  for (int i = threadIdx.x; i < HCC; i += blockDim.x) { ls[i] = 0.f; lq[i] = 0.f; }
  __syncthreads();
  long base = (long)blockIdx.x * BN_ROWS;
  long tot  = (long)BN_ROWS * HCC;
  for (long i = threadIdx.x; i < tot; i += blockDim.x) {
    long row = base + i / HCC; int ch = (int)(i % HCC);
    if (row < n) {
      float v = h[row * HCC + ch];
      atomicAdd(&ls[ch], v);         // ds_add_f32
      atomicAdd(&lq[ch], v * v);
    }
  }
  __syncthreads();
  for (int i = threadIdx.x; i < HCC; i += blockDim.x) {
    atomicAdd(gsum + i, ls[i]);
    atomicAdd(gsq  + i, lq[i]);
  }
}

__global__ void k_bn_final(const float* __restrict__ gsum, const float* __restrict__ gsq,
                           const float* __restrict__ gamma, const float* __restrict__ beta,
                           float* __restrict__ sc, float* __restrict__ sh, long n) {
  int i = blockIdx.x * blockDim.x + threadIdx.x;
  if (i >= HCC) return;
  float inv = 1.0f / (float)n;
  float mu  = gsum[i] * inv;
  float var = gsq[i] * inv - mu * mu;
  float rs  = rsqrtf(var + 1e-5f);
  float scale = gamma[i] * rs;
  sc[i] = scale;
  sh[i] = beta[i] - mu * scale;
}

// h <- leakyrelu(h*sc+sh, 0.01); optionally also write f16 into Ah[N][K2PAD]
__global__ void k_bn_apply(float* __restrict__ h, const float* __restrict__ sc,
                           const float* __restrict__ sh, _Float16* __restrict__ Ah,
                           int writeF16, long n) {
  long i = (long)blockIdx.x * blockDim.x + threadIdx.x;
  if (i >= n * HCC) return;
  int ch = (int)(i % HCC);
  float v = h[i] * sc[ch] + sh[ch];
  v = (v > 0.f) ? v : 0.01f * v;
  h[i] = v;
  if (writeF16) {
    long row = i / HCC;
    Ah[row * K2PAD + ch] = (_Float16)v;
  }
}

// ---------------- classifier: out[n,2] = h[n,:] @ Wc^T + bc ----------------
__global__ void k_cls(const float* __restrict__ h, const float* __restrict__ Wc,
                      const float* __restrict__ bc, float* __restrict__ out, long n) {
  long row = (long)blockIdx.x * blockDim.x + threadIdx.x;
  if (row >= n) return;
  const float* pr = h + (size_t)row * HCC;
  float a0 = bc[0], a1 = bc[1];
  for (int c = 0; c < HCC; ++c) {
    float v = pr[c];
    a0 += v * Wc[c];
    a1 += v * Wc[HCC + c];
  }
  out[row * 2 + 0] = a0;
  out[row * 2 + 1] = a1;
}

// ---------------- host launcher ----------------
extern "C" void kernel_launch(void* const* d_in, const int* in_sizes, int n_in,
                              void* d_out, int out_size, void* d_ws, size_t ws_size,
                              hipStream_t stream) {
  const float* x    = (const float*)d_in[0];
  const int*   ei   = (const int*)d_in[1];   // [2,E] int32
  const float* Wl1  = (const float*)d_in[2];
  const float* bl1  = (const float*)d_in[3];
  const float* Wr1  = (const float*)d_in[4];
  const float* br1  = (const float*)d_in[5];
  const float* att1 = (const float*)d_in[6];
  const float* Wl2  = (const float*)d_in[8];
  const float* bl2  = (const float*)d_in[9];
  const float* Wr2  = (const float*)d_in[10];
  const float* br2  = (const float*)d_in[11];
  const float* att2 = (const float*)d_in[12];
  const float* gamma = (const float*)d_in[14];
  const float* beta  = (const float*)d_in[15];
  const float* Wc   = (const float*)d_in[16];
  const float* bc   = (const float*)d_in[17];
  float* out = (float*)d_out;

  const long N = in_sizes[0] / 128;   // F_IN = 128
  const long E = in_sizes[1] / 2;

  // workspace carve-out
  char* p = (char*)d_ws;
  auto carve = [&](size_t bytes) { char* r = p; p += (bytes + 255) & ~(size_t)255; return r; };
  _Float16* Ah  = (_Float16*)carve((size_t)N * K2PAD * 2);  // f16 activations (lda=128 L1, 320 L2)
  float* xl     = (float*)carve((size_t)N * HCC * 4);
  float* xr     = (float*)carve((size_t)N * HCC * 4);
  float* hb     = (float*)carve((size_t)N * HCC * 4);       // GAT accumulator / BN buffer
  float* sbuf   = (float*)carve((size_t)E * HEADS * 4);
  unsigned* mkey = (unsigned*)carve((size_t)N * HEADS * 4);
  float* den    = (float*)carve((size_t)N * HEADS * 4);
  _Float16* Wlh = (_Float16*)carve((size_t)NPADR * K2PAD * 2);
  _Float16* Wrh = (_Float16*)carve((size_t)NPADR * K2PAD * 2);
  float* stats  = (float*)carve((size_t)4 * HCC * 4);
  float* gsum = stats, *gsq = stats + HCC, *sc = stats + 2 * HCC, *sh = stats + 3 * HCC;

  auto nblk = [](long n, int b) { return dim3((unsigned)((n + b - 1) / b)); };

  const long mtiles = (N + 15) / 16;
  const long gemm_waves = mtiles * (NPADR / 16);
  dim3 gemm_grid = nblk(gemm_waves * 32, 256);

  auto run_gat = [&](const float* Wl, const float* blv, const float* Wr, const float* brv,
                     const float* att, int Kin, int Kpad) {
    long wtot = (long)NPADR * Kpad;
    k_cvt_w<<<nblk(wtot, 256), 256, 0, stream>>>(Wl, Wlh, HCC, Kin, Kpad);
    k_cvt_w<<<nblk(wtot, 256), 256, 0, stream>>>(Wr, Wrh, HCC, Kin, Kpad);
    k_gemm_wmma<<<gemm_grid, 256, 0, stream>>>(Ah, Wlh, blv, xl, (int)N, Kpad, HCC);
    k_gemm_wmma<<<gemm_grid, 256, 0, stream>>>(Ah, Wrh, brv, xr, (int)N, Kpad, HCC);
    k_zero_u32<<<nblk(N * HEADS, 256), 256, 0, stream>>>(mkey, N * HEADS);
    k_zero_u32<<<nblk(N * HEADS, 256), 256, 0, stream>>>((unsigned*)den, N * HEADS);
    k_zero_u32<<<nblk(N * HCC, 256), 256, 0, stream>>>((unsigned*)hb, N * HCC);
    k_edge_score<<<nblk(E * HEADS, 256), 256, 0, stream>>>(ei, xl, xr, att, sbuf, mkey, E);
    k_edge_exp<<<nblk(E * HEADS, 256), 256, 0, stream>>>(ei, sbuf, mkey, den, E);
    k_edge_aggr<<<nblk(E * 32, 256), 256, 0, stream>>>(ei, xl, sbuf, den, hb, E);
    // per-layer bias b1/b2 omitted: it cancels exactly inside BatchNorm (x - mu)
  };

  auto run_bn = [&](int writeF16) {
    k_zero_u32<<<nblk(2 * HCC, 256), 256, 0, stream>>>((unsigned*)gsum, 2 * HCC);
    k_bn_stats<<<nblk((N + BN_ROWS - 1) / BN_ROWS, 1) , 256, 0, stream>>>(hb, gsum, gsq, N);
    k_bn_final<<<nblk(HCC, 256), 256, 0, stream>>>(gsum, gsq, gamma, beta, sc, sh, N);
    if (writeF16)
      k_pad_ah<<<nblk(N * (K2PAD - HCC), 256), 256, 0, stream>>>(Ah, N);
    k_bn_apply<<<nblk(N * HCC, 256), 256, 0, stream>>>(hb, sc, sh, Ah, writeF16, N);
  };

  // layer 1: x (f32,128) -> f16
  k_cvt_f16<<<nblk(N * 128, 256), 256, 0, stream>>>(x, Ah, N * 128);
  run_gat(Wl1, bl1, Wr1, br1, att1, /*Kin=*/128, /*Kpad=*/128);
  run_bn(/*writeF16=*/1);

  // layer 2: Ah is [N][320] f16 (padded)
  run_gat(Wl2, bl2, Wr2, br2, att2, /*Kin=*/HCC, /*Kpad=*/K2PAD);
  run_bn(/*writeF16=*/0);

  // classifier
  k_cls<<<nblk(N, 256), 256, 0, stream>>>(hb, Wc, bc, out, N);
}
